// Bilinear_51350628991117
// MI455X (gfx1250) — compile-verified, matrix operand order checked by reading, not run on hardware
//
#include <hip/hip_runtime.h>

typedef __attribute__((ext_vector_type(16))) __bf16 v16bf;
typedef __attribute__((ext_vector_type(2)))  __bf16 v2bf;
typedef __attribute__((ext_vector_type(8)))  float  v8f;

// Problem sizes (fixed by the reference).
#define B_ROWS 2048
#define KD     512     // k dimension of weight (H,K,L)
#define LD     512     // l dimension
#define HD     512     // output features

// Tiling
#define BM     128     // output rows per workgroup (8 waves x 16)
#define BN      64     // output cols per workgroup (4 WMMA frags per wave)
#define KSTAGE   8     // k-slices staged in LDS per barrier
#define LROW    40     // LDS row pitch in ushorts (32 data + 8 pad)

// Pack two f32 into packed bf16x2 (low = first arg).
__device__ __forceinline__ unsigned pack2bf(float lo, float hi) {
#if __has_builtin(__builtin_amdgcn_cvt_pk_bf16_f32)
    union { v2bf v; unsigned u; } r;
    r.v = __builtin_amdgcn_cvt_pk_bf16_f32(lo, hi);
    return r.u;
#else
    // round-half-up then truncate via byte permute (1 v_perm + 2 v_add,
    // and the adds dual-issue with surrounding muls)
    unsigned ulo = __float_as_uint(lo) + 0x8000u;
    unsigned uhi = __float_as_uint(hi) + 0x8000u;
    return __builtin_amdgcn_perm(uhi, ulo, 0x07060302u);
#endif
}

// Packed bf16x2 multiply -> V_PK_MUL_BF16
__device__ __forceinline__ unsigned pkmul_bf16(unsigned x, unsigned y) {
    v2bf r = __builtin_bit_cast(v2bf, x) * __builtin_bit_cast(v2bf, y);
    return __builtin_bit_cast(unsigned, r);
}

union BFFrag {
    v16bf v;
    unsigned int u[8];
    uint4 q[2];
};

__global__ __launch_bounds__(256)
void bilinear_wmma_bf16(const float* __restrict__ a,
                        const float* __restrict__ b,
                        const float* __restrict__ w,
                        const float* __restrict__ bias,
                        float* __restrict__ out)
{
    // bf16 weight tile: [KSTAGE][BN j][32 l (+pad)]
    __shared__ __align__(16) unsigned short sB[KSTAGE * BN * LROW];

    const int t     = threadIdx.x;
    const int lane  = t & 31;
    const int wave  = t >> 5;      // 0..7 : which 16-row M sub-tile
    const int laneM = lane & 15;   // A-frag row / B-frag column / C-frag N
    const int laneH = lane >> 4;   // K-group selector (and C-frag M-half)

    const int mtile = blockIdx.x * BM;
    const int ntile = blockIdx.y * BN;

    // Row of A (and of a/b) this lane supplies for the A-fragment.
    const int arow = mtile + wave * 16 + laneM;

    // Cooperative weight staging mapping: 4 threads per j row, 8 l each.
    const int sj = t >> 2;                  // 0..63 (j within N tile)
    const int sq = t & 3;                   // 0..3  (which 8-wide l slice)
    const size_t KL = (size_t)KD * LD;
    const float* wrow = w + (size_t)(ntile + sj) * KL + (size_t)sq * 8;

    // Per-lane constant parts of LDS addresses (offsets folded as immediates).
    unsigned short* sSt = &sB[sj * LROW + sq * 8];             // staging store base
    const unsigned short* sLd = &sB[laneM * LROW + laneH * 8]; // frag load base

    v8f acc[4];
#pragma unroll
    for (int nb = 0; nb < 4; ++nb)
#pragma unroll
        for (int r = 0; r < 8; ++r) acc[nb][r] = 0.0f;

    // Loop over 32-wide l-chunks of the (k,l) reduction; the per-lane b slice
    // is converted to packed bf16 ONCE and reused for 512 consecutive k steps.
    for (int lc = 0; lc < LD / 32; ++lc) {
        const int l0 = lc * 32;

        // Per-lane b values matching the 16-bit A-matrix VGPR layout:
        // elements 0..7  -> l = l0 + laneH*8 + i
        // elements 8..15 -> l = l0 + 16 + laneH*8 + i
        BFFrag bvbf;
        {
            const float* bp = b + (size_t)arow * LD + l0 + laneH * 8;
            float4 b0 = *(const float4*)(bp);
            float4 b1 = *(const float4*)(bp + 4);
            float4 b2 = *(const float4*)(bp + 16);
            float4 b3 = *(const float4*)(bp + 20);
            bvbf.u[0] = pack2bf(b0.x, b0.y);
            bvbf.u[1] = pack2bf(b0.z, b0.w);
            bvbf.u[2] = pack2bf(b1.x, b1.y);
            bvbf.u[3] = pack2bf(b1.z, b1.w);
            bvbf.u[4] = pack2bf(b2.x, b2.y);
            bvbf.u[5] = pack2bf(b2.z, b2.w);
            bvbf.u[6] = pack2bf(b3.x, b3.y);
            bvbf.u[7] = pack2bf(b3.z, b3.w);
        }

        const float* wlc = wrow + l0;

        for (int k0 = 0; k0 < KD; k0 += KSTAGE) {
            __syncthreads();   // previous tile fully consumed

            // ---- Stage KSTAGE k-slices of the weight tile as bf16 ----
#pragma unroll
            for (int kk = 0; kk < KSTAGE; ++kk) {
                const float* wp = wlc + (size_t)(k0 + kk) * LD;
                float4 w0 = *(const float4*)(wp);
                float4 w1 = *(const float4*)(wp + 4);
                uint4 u;
                u.x = pack2bf(w0.x, w0.y);
                u.y = pack2bf(w0.z, w0.w);
                u.z = pack2bf(w1.x, w1.y);
                u.w = pack2bf(w1.z, w1.w);
                *(uint4*)(sSt + kk * (BN * LROW)) = u;
            }
            if (k0 + KSTAGE < KD) {
                // hint the next stage into cache (global_prefetch_b8)
                __builtin_prefetch(wlc + (size_t)(k0 + KSTAGE) * LD, 0, 1);
            }

            // a[arow, k0..k0+7] for this lane's row
            const float* ap = a + (size_t)arow * KD + k0;
            float4 a0 = *(const float4*)(ap);
            float4 a1 = *(const float4*)(ap + 4);
            float areg[8] = {a0.x, a0.y, a0.z, a0.w, a1.x, a1.y, a1.z, a1.w};

            __syncthreads();   // tile visible to all waves

            // ---- Compute: 4 WMMAs per k-slice per wave ----
#pragma unroll
            for (int kk = 0; kk < KSTAGE; ++kk) {
                // splat a[arow,k] as packed bf16x2 (2 VALU)
                const unsigned sbits = pack2bf(areg[kk], areg[kk]);
                // A-frag = sbits * bvbf : 8x v_pk_mul_bf16
                BFFrag af;
#pragma unroll
                for (int i = 0; i < 8; ++i) {
                    af.u[i] = pkmul_bf16(sbits, bvbf.u[i]);
                }
#pragma unroll
                for (int nb = 0; nb < 4; ++nb) {
                    const unsigned short* base =
                        sLd + kk * (BN * LROW) + nb * (16 * LROW);
                    BFFrag bf;
                    bf.q[0] = *(const uint4*)(base);        // l = laneH*8 + 0..7
                    bf.q[1] = *(const uint4*)(base + 16);   // l = 16 + laneH*8 + 0..7
                    acc[nb] = __builtin_amdgcn_wmma_f32_16x16x32_bf16(
                        false, af.v, false, bf.v,
                        (short)0, acc[nb], false, false);
                }
            }
        }
    }

    // ---- Epilogue: C layout lane=N, (laneH*8 + VGPR)=M; fold in bias ----
#pragma unroll
    for (int nb = 0; nb < 4; ++nb) {
        const int col = ntile + nb * 16 + laneM;
        const float bz = bias[col];
#pragma unroll
        for (int r = 0; r < 8; ++r) {
            const int orow = mtile + wave * 16 + laneH * 8 + r;
            out[(size_t)orow * HD + col] = acc[nb][r] + bz;
        }
    }
}

extern "C" void kernel_launch(void* const* d_in, const int* in_sizes, int n_in,
                              void* d_out, int out_size, void* d_ws, size_t ws_size,
                              hipStream_t stream) {
    const float* a    = (const float*)d_in[0];   // (2048, 512)
    const float* b    = (const float*)d_in[1];   // (2048, 512)
    const float* wgt  = (const float*)d_in[2];   // (512, 512, 512)
    const float* bias = (const float*)d_in[3];   // (512,)
    float* out        = (float*)d_out;           // (2048, 512)

    dim3 grid(B_ROWS / BM, HD / BN);             // 16 x 8 = 128 workgroups
    dim3 block(256);
    bilinear_wmma_bf16<<<grid, block, 0, stream>>>(a, b, wgt, bias, out);
}
